// NonLocalBlock_12421045420655
// MI455X (gfx1250) — matrix operand order repaired and checked by reading, not compiled, not run on hardware
//
#include <hip/hip_runtime.h>
#include <hip/hip_bf16.h>

typedef __bf16 bf16_t;
typedef __attribute__((ext_vector_type(16))) __bf16 v16bf;
typedef __attribute__((ext_vector_type(8)))  __bf16 v8bf;
typedef __attribute__((ext_vector_type(8)))  float  v8f;
typedef int v4i_t __attribute__((vector_size(16)));

#define BM 128
#define BN 128
#define BK 64
#define LDSA 72   // padded LDS row (144B = 36 banks -> conflict-free b128 lane pattern)

// ---- CDNA5 async global->LDS copy path (probe-confirmed builtin) ------------
#if defined(__has_builtin)
#  if __has_builtin(__builtin_amdgcn_global_load_async_to_lds_b128)
#    define HAVE_ASYNC 1
#  endif
#endif
#ifndef HAVE_ASYNC
#  define HAVE_ASYNC 0
#endif

#if HAVE_ASYNC
static __device__ __forceinline__ void async_cp16(const void* g, void* l) {
    __builtin_amdgcn_global_load_async_to_lds_b128(
        (__attribute__((address_space(1))) v4i_t*)(unsigned long long)g,
        (__attribute__((address_space(3))) v4i_t*)(unsigned int)(unsigned long long)l,
        0, 0);
}
#endif

// ---------------------------------------------------------------------------
// Generic bf16 WMMA GEMM:  C[M,N] = A[M,K] * B + (bias) + (residual)
//   BLAY = 0 : B stored [N][K] (K contiguous, "NT")
//   BLAY = 1 : B stored [K][N] (N contiguous, "NN") -> transposed while staging
//   BIAS = 0 none, 1 per-M, 2 per-N ;  RES: add f32 residual (C layout, ld=LDC)
//   OUTM = 0 f32 out, 1 bf16 out ;  LDC: compile-time output leading dim
// 256 threads = 8 waves (4x2); macro tile 128x128; wave tile 32x64;
// double-buffered LDS, BK=64 (2 WMMA K-steps, 16 WMMAs per stage).
// M,N multiples of 128; K multiple of 64 (true for all calls below).
// ---------------------------------------------------------------------------
template<int BLAY, int BIAS, bool RES, int OUTM, int LDC>
__global__ __launch_bounds__(256) void gemm_bf16_wmma(
    const bf16_t* __restrict__ Ag, long strideA, int lda,
    const bf16_t* __restrict__ Bg, long strideB, int ldb,
    float*  __restrict__ Cf, long strideC,
    bf16_t* __restrict__ Cb, long strideCb,
    const float* __restrict__ bias,
    const float* __restrict__ resid, long strideR,
    int K)
{
    __shared__ bf16_t sA[2][BM * LDSA];
    __shared__ bf16_t sB[2][BN * LDSA];

    const int  bz   = blockIdx.z;
    const int  tid  = threadIdx.x;
    const int  lane = tid & 31;
    const int  wave = tid >> 5;
    const int  wr   = wave & 3;          // 32-row group
    const int  wc   = wave >> 2;         // 64-col group
    const long m0   = (long)blockIdx.y * BM;
    const long n0   = (long)blockIdx.x * BN;

    const bf16_t* A  = Ag + (long)bz * strideA;
    const bf16_t* Bm = Bg + (long)bz * strideB;

    v8f acc[2][4];
    #pragma unroll
    for (int i = 0; i < 2; ++i)
        #pragma unroll
        for (int j = 0; j < 4; ++j)
            #pragma unroll
            for (int e = 0; e < 8; ++e) acc[i][j][e] = 0.0f;

    const int r  = lane & 15;
    const int hs = lane >> 4;
    const int nt = K / BK;

// ---- staging / compute macros ---------------------------------------------
#define LOAD_A_IDX(c_)  int row_ = (c_) >> 3, kc_ = (c_) & 7
#define GA(KT)          (A  + (m0 + row_) * (long)lda + (KT) * BK + kc_ * 8)
#define GB_NT(KT)       (Bm + (n0 + row_) * (long)ldb + (KT) * BK + kc_ * 8)
#define LA(BUF)         (&sA[BUF][row_ * LDSA + kc_ * 8])
#define LB(BUF)         (&sB[BUF][row_ * LDSA + kc_ * 8])

#define STAGE_B_SYNC_NN(KT, BUF) do {                                          \
    _Pragma("unroll")                                                          \
    for (int it_ = 0; it_ < 4; ++it_) {                                        \
        int c_ = tid + it_ * 256;                                              \
        int kr_ = c_ >> 4, nc_ = c_ & 15;                                      \
        v8bf v_ = *(const v8bf*)(Bm + ((long)(KT) * BK + kr_) * ldb + n0 + nc_ * 8); \
        _Pragma("unroll")                                                      \
        for (int j_ = 0; j_ < 8; ++j_)                                         \
            sB[BUF][(nc_ * 8 + j_) * LDSA + kr_] = v_[j_];                     \
    } } while (0)

#define PREFETCH(KT) do {                                                      \
    if ((KT) + 2 < nt) {                                                       \
        if (tid < 128) {                                                       \
            __builtin_prefetch(A + (m0 + tid) * (long)lda + ((KT) + 2) * BK, 0, 1); \
            if (BLAY == 0)                                                     \
                __builtin_prefetch(Bm + (n0 + tid) * (long)ldb + ((KT) + 2) * BK, 0, 1); \
        }                                                                      \
        if (BLAY == 1 && tid < 64)                                             \
            __builtin_prefetch(Bm + ((long)((KT) + 2) * BK + tid) * ldb + n0, 0, 1); \
    } } while (0)

#define COMPUTE(BUF) do {                                                      \
    _Pragma("unroll")                                                          \
    for (int kk_ = 0; kk_ < 2; ++kk_) {                                        \
        const int kaof_ = kk_ * 32 + hs * 8;                                   \
        const int kbof_ = kk_ * 32 + hs * 8;                                   \
        v16bf af_[2]; v16bf bq_[4];                                            \
        _Pragma("unroll")                                                      \
        for (int i_ = 0; i_ < 2; ++i_) {                                       \
            const bf16_t* p_ = &sA[BUF][(wr * 32 + i_ * 16 + r) * LDSA + kaof_]; \
            v8bf lo_ = *(const v8bf*)(p_);                                     \
            v8bf hi_ = *(const v8bf*)(p_ + 16);                                \
            af_[i_] = __builtin_shufflevector(lo_, hi_,                        \
                       0,1,2,3,4,5,6,7,8,9,10,11,12,13,14,15);                 \
        }                                                                      \
        _Pragma("unroll")                                                      \
        for (int j_ = 0; j_ < 4; ++j_) {                                       \
            const bf16_t* p_ = &sB[BUF][(wc * 64 + j_ * 16 + r) * LDSA + kbof_]; \
            v8bf lo_ = *(const v8bf*)(p_);                                     \
            v8bf hi_ = *(const v8bf*)(p_ + 16);                                \
            bq_[j_] = __builtin_shufflevector(lo_, hi_,                        \
                       0,1,2,3,4,5,6,7,8,9,10,11,12,13,14,15);                 \
        }                                                                      \
        _Pragma("unroll")                                                      \
        for (int i_ = 0; i_ < 2; ++i_)                                         \
            _Pragma("unroll")                                                  \
            for (int j_ = 0; j_ < 4; ++j_)                                     \
                acc[i_][j_] = __builtin_amdgcn_wmma_f32_16x16x32_bf16(         \
                    false, af_[i_], false, bq_[j_], (short)0, acc[i_][j_],     \
                    false, false);                                             \
    } } while (0)

#if HAVE_ASYNC
    // ---- async-DMA pipelined main loop ----
    #pragma unroll
    for (int it_ = 0; it_ < 4; ++it_) { LOAD_A_IDX(tid + it_ * 256);
        async_cp16(GA(0), LA(0));
        if (BLAY == 0) async_cp16(GB_NT(0), LB(0));
    }
    if (BLAY == 1) STAGE_B_SYNC_NN(0, 0);
    asm volatile("s_wait_asynccnt 0x0" ::: "memory");
    __syncthreads();
    for (int kt = 0; kt < nt; ++kt) {
        const int cur = kt & 1;
        PREFETCH(kt);
        if (kt + 1 < nt) {
            #pragma unroll
            for (int it_ = 0; it_ < 4; ++it_) { LOAD_A_IDX(tid + it_ * 256);
                async_cp16(GA(kt + 1), LA(cur ^ 1));
                if (BLAY == 0) async_cp16(GB_NT(kt + 1), LB(cur ^ 1));
            }
            if (BLAY == 1) STAGE_B_SYNC_NN(kt + 1, cur ^ 1);
        }
        COMPUTE(cur);
        if (kt + 1 < nt) asm volatile("s_wait_asynccnt 0x0" ::: "memory");
        __syncthreads();
    }
#else
    // ---- register-pipelined fallback ----
    v8bf ra[4], rb[4];
    #pragma unroll
    for (int it_ = 0; it_ < 4; ++it_) { LOAD_A_IDX(tid + it_ * 256);
        ra[it_] = *(const v8bf*)GA(0);
        if (BLAY == 0) rb[it_] = *(const v8bf*)GB_NT(0);
    }
    #pragma unroll
    for (int it_ = 0; it_ < 4; ++it_) { LOAD_A_IDX(tid + it_ * 256);
        *(v8bf*)LA(0) = ra[it_];
        if (BLAY == 0) *(v8bf*)LB(0) = rb[it_];
    }
    if (BLAY == 1) STAGE_B_SYNC_NN(0, 0);
    __syncthreads();
    for (int kt = 0; kt < nt; ++kt) {
        const int cur = kt & 1;
        PREFETCH(kt);
        if (kt + 1 < nt) {
            #pragma unroll
            for (int it_ = 0; it_ < 4; ++it_) { LOAD_A_IDX(tid + it_ * 256);
                ra[it_] = *(const v8bf*)GA(kt + 1);
                if (BLAY == 0) rb[it_] = *(const v8bf*)GB_NT(kt + 1);
            }
        }
        COMPUTE(cur);
        if (kt + 1 < nt) {
            #pragma unroll
            for (int it_ = 0; it_ < 4; ++it_) { LOAD_A_IDX(tid + it_ * 256);
                *(v8bf*)LA(cur ^ 1) = ra[it_];
                if (BLAY == 0) *(v8bf*)LB(cur ^ 1) = rb[it_];
            }
            if (BLAY == 1) STAGE_B_SYNC_NN(kt + 1, cur ^ 1);
        }
        __syncthreads();
    }
#endif
#undef LOAD_A_IDX
#undef GA
#undef GB_NT
#undef LA
#undef LB
#undef STAGE_B_SYNC_NN
#undef PREFETCH
#undef COMPUTE

    // ---- epilogue: C/D layout — lane<16: M=v,N=lane ; lane>=16: M=v+8 ------
    const int col   = lane & 15;
    const int rbase = hs * 8;
    #pragma unroll
    for (int i = 0; i < 2; ++i) {
        #pragma unroll
        for (int j = 0; j < 4; ++j) {
            long gm = m0 + wr * 32 + i * 16 + rbase;
            long gn = n0 + wc * 64 + j * 16 + col;
            #pragma unroll
            for (int v = 0; v < 8; ++v) {
                long row = gm + v;
                float val = acc[i][j][v];
                if (BIAS == 1) val += bias[row];
                else if (BIAS == 2) val += bias[gn];
                if (RES) val += resid[(long)bz * strideR + row * LDC + gn];
                if (OUTM == 0) Cf[(long)bz * strideC  + row * LDC + gn] = val;
                else           Cb[(long)bz * strideCb + row * LDC + gn] = (bf16_t)val;
            }
        }
    }
}

// ---------------------------------------------------------------------------
// x [B][C][T] f32  ->  xT [B][T][C] bf16   (tiled transpose + convert)
// ---------------------------------------------------------------------------
__global__ __launch_bounds__(256) void transpose_cvt_kernel(
    const float* __restrict__ x, bf16_t* __restrict__ xT, int C, int T)
{
    __shared__ float tile[32][33];
    const int b  = blockIdx.z;
    const int c0 = blockIdx.y * 32;
    const int t0 = blockIdx.x * 32;
    const int tx = threadIdx.x;      // 32
    const int ty = threadIdx.y;      // 8
    #pragma unroll
    for (int i = 0; i < 32; i += 8)
        tile[ty + i][tx] = x[(long)b * C * T + (long)(c0 + ty + i) * T + t0 + tx];
    __syncthreads();
    #pragma unroll
    for (int i = 0; i < 32; i += 8)
        xT[(long)b * T * C + (long)(t0 + ty + i) * C + c0 + tx] =
            (bf16_t)tile[tx][ty + i];
}

// f32 -> bf16 elementwise (weights)
__global__ void cvt_f32_bf16_kernel(const float* __restrict__ in,
                                    bf16_t* __restrict__ out, int n)
{
    int i = blockIdx.x * blockDim.x + threadIdx.x;
    if (i < n) out[i] = (bf16_t)in[i];
}

// ---------------------------------------------------------------------------
// In-place row softmax: reads 2048 f32, writes 2048 bf16 packed at row start.
// ---------------------------------------------------------------------------
__global__ __launch_bounds__(256) void softmax_rows_kernel(float* __restrict__ scores, int T)
{
    const long row = blockIdx.x;
    float* p  = scores + row * (long)T;
    bf16_t* o = (bf16_t*)p;
    const int tid = threadIdx.x;

    float v[8];
    float mx = -3.402823466e+38f;
    #pragma unroll
    for (int i = 0; i < 8; ++i) { v[i] = p[tid + i * 256]; mx = fmaxf(mx, v[i]); }

    __shared__ float red[256];
    red[tid] = mx; __syncthreads();
    for (int s = 128; s > 0; s >>= 1) {
        if (tid < s) red[tid] = fmaxf(red[tid], red[tid + s]);
        __syncthreads();
    }
    mx = red[0]; __syncthreads();

    float sum = 0.f;
    #pragma unroll
    for (int i = 0; i < 8; ++i) { v[i] = __expf(v[i] - mx); sum += v[i]; }
    red[tid] = sum; __syncthreads();
    for (int s = 128; s > 0; s >>= 1) {
        if (tid < s) red[tid] += red[tid + s];
        __syncthreads();
    }
    const float inv = 1.0f / red[0];
    __syncthreads();
    #pragma unroll
    for (int i = 0; i < 8; ++i) o[tid + i * 256] = (bf16_t)(v[i] * inv);
}

// ---------------------------------------------------------------------------
// BatchNorm: per-channel mean/var over (B,T), then normalize.
// ---------------------------------------------------------------------------
__global__ __launch_bounds__(256) void bn_stats_kernel(
    const float* __restrict__ s, float* __restrict__ mean, float* __restrict__ var,
    int B, int C, int T)
{
    const int c   = blockIdx.x;
    const int tid = threadIdx.x;
    float sum = 0.f, sq = 0.f;
    const long n = (long)B * T;
    for (long i = tid; i < n; i += 256) {
        int b = (int)(i / T);
        int t = (int)(i % T);
        float vv = s[(long)b * C * T + (long)c * T + t];
        sum += vv; sq += vv * vv;
    }
    __shared__ float rs[256], rq[256];
    rs[tid] = sum; rq[tid] = sq; __syncthreads();
    for (int st = 128; st > 0; st >>= 1) {
        if (tid < st) { rs[tid] += rs[tid + st]; rq[tid] += rq[tid + st]; }
        __syncthreads();
    }
    if (tid == 0) {
        float m = rs[0] / (float)n;
        mean[c] = m;
        var[c]  = rq[0] / (float)n - m * m;
    }
}

__global__ void bn_apply_kernel(
    const float* __restrict__ s, const float* __restrict__ mean,
    const float* __restrict__ var, const float* __restrict__ gamma,
    const float* __restrict__ beta, float* __restrict__ out,
    int C, int T, long total)
{
    long idx = (long)blockIdx.x * blockDim.x + threadIdx.x;
    if (idx >= total) return;
    long rem = idx % ((long)C * T);
    int  c   = (int)(rem / T);
    out[idx] = gamma[c] * (s[idx] - mean[c]) * __frsqrt_rn(var[c] + 1e-5f) + beta[c];
}

// ---------------------------------------------------------------------------
extern "C" void kernel_launch(void* const* d_in, const int* in_sizes, int n_in,
                              void* d_out, int out_size, void* d_ws, size_t ws_size,
                              hipStream_t stream) {
    (void)in_sizes; (void)n_in; (void)out_size; (void)ws_size;
    constexpr int  Bb = 8, C = 1024, T = 2048, I = 512;

    const float* x       = (const float*)d_in[0];
    const float* g_w     = (const float*)d_in[1];
    const float* g_b     = (const float*)d_in[2];
    const float* theta_w = (const float*)d_in[3];
    const float* theta_b = (const float*)d_in[4];
    const float* phi_w   = (const float*)d_in[5];
    const float* phi_b   = (const float*)d_in[6];
    const float* W_w     = (const float*)d_in[7];
    const float* W_b     = (const float*)d_in[8];
    const float* gamma   = (const float*)d_in[9];
    const float* beta    = (const float*)d_in[10];
    // d_in[11] = epoch (unused by the math)

    char* ws = (char*)d_ws;
    size_t off = 0;
    auto carve = [&](size_t bytes) { size_t o = off; off = (off + bytes + 255) & ~(size_t)255; return o; };

    size_t off_xT  = carve((size_t)Bb * T * C * sizeof(bf16_t));   // reused for y
    size_t off_wg  = carve((size_t)I * C * sizeof(bf16_t));
    size_t off_wth = carve((size_t)I * C * sizeof(bf16_t));
    size_t off_wph = carve((size_t)I * C * sizeof(bf16_t));
    size_t off_wW  = carve((size_t)C * I * sizeof(bf16_t));
    size_t off_th  = carve((size_t)Bb * T * I * sizeof(bf16_t));
    size_t off_ph  = carve((size_t)Bb * T * I * sizeof(bf16_t));
    size_t off_g   = carve((size_t)Bb * T * I * sizeof(bf16_t));
    size_t off_sc  = carve((size_t)Bb * T * T * sizeof(float));    // scores; reused for s
    size_t off_mu  = carve((size_t)C * sizeof(float));
    size_t off_var = carve((size_t)C * sizeof(float));

    bf16_t* xT    = (bf16_t*)(ws + off_xT);
    bf16_t* wg    = (bf16_t*)(ws + off_wg);
    bf16_t* wth   = (bf16_t*)(ws + off_wth);
    bf16_t* wph   = (bf16_t*)(ws + off_wph);
    bf16_t* wW    = (bf16_t*)(ws + off_wW);
    bf16_t* thb   = (bf16_t*)(ws + off_th);
    bf16_t* phb   = (bf16_t*)(ws + off_ph);
    bf16_t* gb    = (bf16_t*)(ws + off_g);
    float*  sc    = (float*)(ws + off_sc);
    bf16_t* yb    = (bf16_t*)(ws + off_xT);   // reuse xT region after projections
    float*  sbuf  = (float*)(ws + off_sc);    // reuse score region after attn GEMM
    float*  mu    = (float*)(ws + off_mu);
    float*  var   = (float*)(ws + off_var);

    // 1) weights f32 -> bf16
    cvt_f32_bf16_kernel<<<(I * C + 255) / 256, 256, 0, stream>>>(g_w,     wg,  I * C);
    cvt_f32_bf16_kernel<<<(I * C + 255) / 256, 256, 0, stream>>>(theta_w, wth, I * C);
    cvt_f32_bf16_kernel<<<(I * C + 255) / 256, 256, 0, stream>>>(phi_w,   wph, I * C);
    cvt_f32_bf16_kernel<<<(C * I + 255) / 256, 256, 0, stream>>>(W_w,     wW,  C * I);

    // 2) x [B][C][T] -> xT [B][T][C] bf16
    transpose_cvt_kernel<<<dim3(T / 32, C / 32, Bb), dim3(32, 8), 0, stream>>>(x, xT, C, T);

    // 3) projections: out[t][i] = sum_c xT[t][c] * w[i][c] + b[i]  (NT, bias per-N, bf16 out)
    {
        dim3 grid(I / BN, T / BM, Bb);
        gemm_bf16_wmma<0, 2, false, 1, 512><<<grid, 256, 0, stream>>>(
            xT, (long)T * C, C,  wth, 0, C,
            nullptr, 0,  thb, (long)T * I,
            theta_b, nullptr, 0, C);
        gemm_bf16_wmma<0, 2, false, 1, 512><<<grid, 256, 0, stream>>>(
            xT, (long)T * C, C,  wph, 0, C,
            nullptr, 0,  phb, (long)T * I,
            phi_b, nullptr, 0, C);
        gemm_bf16_wmma<0, 2, false, 1, 512><<<grid, 256, 0, stream>>>(
            xT, (long)T * C, C,  wg, 0, C,
            nullptr, 0,  gb, (long)T * I,
            g_b, nullptr, 0, C);
    }

    // 4) scores f[t][s] = theta[t][:] . phi[s][:]   (NT, f32 out, LDC = T)
    gemm_bf16_wmma<0, 0, false, 0, 2048><<<dim3(T / BN, T / BM, Bb), 256, 0, stream>>>(
        thb, (long)T * I, I,  phb, (long)T * I, I,
        sc, (long)T * T,  nullptr, 0,
        nullptr, nullptr, 0, I);

    // 5) softmax over last axis, in place (bf16 packed at each row start)
    softmax_rows_kernel<<<Bb * T, 256, 0, stream>>>(sc, T);

    // 6) y[t][i] = sum_s attn[t][s] * g[s][i]   (A = bf16 view of score rows, B NN, bf16 out)
    gemm_bf16_wmma<1, 0, false, 1, 512><<<dim3(I / BN, T / BM, Bb), 256, 0, stream>>>(
        (const bf16_t*)sc, 2L * T * T, 2 * T,  gb, (long)T * I, I,
        nullptr, 0,  yb, (long)T * I,
        nullptr, nullptr, 0, T);

    // 7) s[c][t] = sum_i W[c][i] * y[t][i] + W_b[c] + x[c][t]   (NT, bias per-M, residual, f32)
    gemm_bf16_wmma<0, 1, true, 0, 2048><<<dim3(T / BN, C / BM, Bb), 256, 0, stream>>>(
        wW, 0, I,  yb, (long)T * I, I,
        sbuf, (long)C * T,  nullptr, 0,
        W_b, x, (long)C * T, I);

    // 8) BatchNorm
    bn_stats_kernel<<<C, 256, 0, stream>>>(sbuf, mu, var, Bb, C, T);
    long total = (long)Bb * C * T;
    bn_apply_kernel<<<(unsigned)((total + 255) / 256), 256, 0, stream>>>(
        sbuf, mu, var, gamma, beta, (float*)d_out, C, T, total);
}
